// Attention_Module_51702816309882
// MI455X (gfx1250) — matrix operand order repaired and checked by reading, not compiled, not run on hardware
//
#include <hip/hip_runtime.h>

typedef __attribute__((ext_vector_type(16))) __bf16 v16bf;
typedef __attribute__((ext_vector_type(8)))  float  v8f;
typedef __attribute__((ext_vector_type(4)))  float  v4f;

#define C_TOT 4096
#define HW    784      // 28*28
#define NB    32       // batch
#define NK    4        // output channels
#define KCH   512      // channels per wave (C_TOT / 8 waves)
#define KSTEPS (KCH / 32)   // 16 WMMA steps per wave, compile-time constant

// ---------------------------------------------------------------------------
// Kernel 1: cams[n][k][hw] = relu( sum_c w[k][c] * x[n][c][hw] )
// Block = 256 threads (8 waves) per (n, 16-pixel tile). Each wave covers a
// 512-channel K-chunk with bf16 hi/lo-split WMMA (3 accumulations per step
// for ~fp32 accuracy), then partials are summed across waves through LDS.
// M = hw tile (16), N = k (4 of 16 used), K = channels.
// ---------------------------------------------------------------------------
__global__ __launch_bounds__(256) void k_cams_wmma(
    const float* __restrict__ x, const float* __restrict__ w,
    float* __restrict__ cams)
{
    const int n      = blockIdx.y;
    const int hw0    = blockIdx.x << 4;          // 16-pixel tile base
    const int tid    = threadIdx.x;
    const int lane   = tid & 31;                 // wave32 lane
    const int wave   = tid >> 5;                 // 0..7
    const int laneLo = lane & 15;
    const int isHi   = lane >> 4;

    // A-matrix (x tile): lane holds row M = laneLo, two contiguous K-runs of 8
    const float* xn  = x + (size_t)n * C_TOT * HW + hw0 + laneLo;
    const int ka0 = isHi ? 8  : 0;   // K for elements 0..7
    const int ka1 = isHi ? 24 : 16;  // K for elements 8..15

    // B-matrix (weights): lane holds column N = laneLo, two contiguous K-runs of 8
    const int kb0   = isHi ? 8 : 0;
    const int brow  = laneLo;                    // N index (k)
    const float bmask = (brow < NK) ? 1.0f : 0.0f;
    const float* wr = w + (size_t)(brow < NK ? brow : 0) * C_TOT;

    v8f acc = {};
    const int cBeg = wave * KCH;
    for (int it = 0; it < KSTEPS; ++it) {        // uniform scalar trip count
        const int c0 = cBeg + (it << 5);
        v16bf a_hi, a_lo, b_hi, b_lo;
        #pragma unroll
        for (int i = 0; i < 8; ++i) {
            float v0 = xn[(size_t)(c0 + ka0 + i) * HW];   // coalesced over lanes
            __bf16 h0 = (__bf16)v0;
            a_hi[i]     = h0;
            a_lo[i]     = (__bf16)(v0 - (float)h0);
            float v1 = xn[(size_t)(c0 + ka1 + i) * HW];
            __bf16 h1 = (__bf16)v1;
            a_hi[i + 8] = h1;
            a_lo[i + 8] = (__bf16)(v1 - (float)h1);
        }
        #pragma unroll
        for (int i = 0; i < 16; ++i) {
            float v = wr[c0 + kb0 + i + (i & 8)] * bmask;           // zero for k>=4 lanes
            __bf16 h = (__bf16)v;
            b_hi[i] = h;
            b_lo[i] = (__bf16)(v - (float)h);
        }
        // D = A*B + C ; hi*hi + hi*lo + lo*hi  (lo*lo ~ 2^-16, negligible)
        acc = __builtin_amdgcn_wmma_f32_16x16x32_bf16(false, a_hi, false, b_hi,
                                                      (short)0, acc, false, false);
        acc = __builtin_amdgcn_wmma_f32_16x16x32_bf16(false, a_hi, false, b_lo,
                                                      (short)0, acc, false, false);
        acc = __builtin_amdgcn_wmma_f32_16x16x32_bf16(false, a_lo, false, b_hi,
                                                      (short)0, acc, false, false);
    }

    // Cross-wave reduction of the 8 partial C tiles through LDS.
    __shared__ float red[8 * 256];
    #pragma unroll
    for (int r = 0; r < 8; ++r)
        red[wave * 256 + lane * 8 + r] = acc[r];
    __syncthreads();

    // Each thread owns one (lane, r) slot of the C tile.
    float v = 0.0f;
    #pragma unroll
    for (int wv = 0; wv < 8; ++wv)
        v += red[wv * 256 + tid];
    const int eLane = tid >> 3;                  // original lane of this slot
    const int eR    = tid & 7;                   // original C element index
    const int eN    = eLane & 15;                // k
    const int eM    = eR + ((eLane >> 4) << 3);  // hw offset within tile
    if (eN < NK)
        cams[((size_t)n * NK + eN) * HW + hw0 + eM] = v > 0.0f ? v : 0.0f;
}

// ---------------------------------------------------------------------------
// Kernel 2: thr[n*4+k] = gama * max_hw(cams[n][k][:])   (128 tiny reductions)
// ---------------------------------------------------------------------------
__global__ __launch_bounds__(256) void k_thr(
    const float* __restrict__ cams, const float* __restrict__ gama,
    float* __restrict__ thr)
{
    const int nk = blockIdx.x;                    // 0..127
    const float* p = cams + (size_t)nk * HW;
    float m = 0.0f;                               // cams >= 0 after relu
    for (int i = threadIdx.x; i < HW; i += 256)
        m = fmaxf(m, p[i]);
    __shared__ float red[256];
    red[threadIdx.x] = m;
    __syncthreads();
    for (int s = 128; s > 0; s >>= 1) {
        if (threadIdx.x < s)
            red[threadIdx.x] = fmaxf(red[threadIdx.x], red[threadIdx.x + s]);
        __syncthreads();
    }
    if (threadIdx.x == 0)
        thr[nk] = red[0] * gama[0];
}

// ---------------------------------------------------------------------------
// Kernel 3: s[n][hw] = 0.25 * sum_k (cams > thr ? 0 : cams)
// ---------------------------------------------------------------------------
__global__ __launch_bounds__(256) void k_spix(
    const float* __restrict__ cams, const float* __restrict__ thr,
    float* __restrict__ s)
{
    const int p = blockIdx.x * 256 + threadIdx.x;
    if (p >= NB * HW) return;
    const int n = p / HW, hw = p % HW;
    float acc = 0.0f;
    #pragma unroll
    for (int k = 0; k < NK; ++k) {
        float cv = cams[((size_t)n * NK + k) * HW + hw];
        float t  = thr[n * NK + k];
        acc += (cv > t) ? 0.0f : cv;
    }
    s[p] = acc * 0.25f;
}

// ---------------------------------------------------------------------------
// Kernel 4: out = x * s  (the bandwidth pass: 822 MB moved, b128 vectorized,
// non-temporal since both streams are touched exactly once here)
// ---------------------------------------------------------------------------
__global__ __launch_bounds__(256) void k_scale(
    const v4f* __restrict__ x4, const float* __restrict__ s,
    v4f* __restrict__ out4)
{
    const int n   = blockIdx.y;
    const int j   = blockIdx.x * 256 + threadIdx.x;   // < 4096*196 = 802816
    const int hw4 = j % 196;
    const v4f sv = *reinterpret_cast<const v4f*>(s + (size_t)n * HW + hw4 * 4);
    const size_t idx = (size_t)n * 802816 + j;
    v4f xv = __builtin_nontemporal_load(&x4[idx]);
    v4f ov = xv * sv;
    __builtin_nontemporal_store(ov, &out4[idx]);
}

extern "C" void kernel_launch(void* const* d_in, const int* in_sizes, int n_in,
                              void* d_out, int out_size, void* d_ws, size_t ws_size,
                              hipStream_t stream)
{
    const float* x    = (const float*)d_in[0];   // (32,4096,28,28)
    const float* w    = (const float*)d_in[1];   // (4,4096,1,1)
    const float* gama = (const float*)d_in[2];   // scalar
    float* out = (float*)d_out;

    char* ws = (char*)d_ws;
    float* cams = (float*)(ws);                  // 32*4*784 f32 = 401408 B
    float* thr  = (float*)(ws + 401408);         // 128 f32     = 512 B
    float* s    = (float*)(ws + 401920);         // 32*784 f32  = 100352 B (16B aligned)

    // 1) skinny GEMM + relu via WMMA: 49 hw-tiles x 32 batches, 8 K-split waves
    k_cams_wmma<<<dim3(49, 32), 256, 0, stream>>>(x, w, cams);
    // 2) per-(n,k) max -> threshold
    k_thr<<<128, 256, 0, stream>>>(cams, gama, thr);
    // 3) per-pixel dropped mean map
    k_spix<<<(NB * HW + 255) / 256, 256, 0, stream>>>(cams, thr, s);
    // 4) bandwidth pass: out = x * s
    k_scale<<<dim3(3136, 32), 256, 0, stream>>>((const v4f*)x, s, (v4f*)out);
}
